// MultiHead3DAttention_65060164600387
// MI455X (gfx1250) — compile-verified, matrix operand order checked by reading, not run on hardware
//
#include <hip/hip_runtime.h>

// ---------------------------------------------------------------------------
// MultiHead 3D attention for MI455X (gfx1250, wave32, WMMA f16->f32 + TDM/async)
// x:[1,512,16,16,16] fp32  -> out:[1,512,16,16,16] fp32
// N=4096 tokens, C=512, heads=8, dh=64, scale = sqrt(64) = 8 (multiplied, per ref)
// ---------------------------------------------------------------------------

typedef __attribute__((ext_vector_type(16))) _Float16 v16h;
typedef __attribute__((ext_vector_type(8)))  float    v8f;
typedef __attribute__((ext_vector_type(4)))  unsigned v4u;
typedef __attribute__((ext_vector_type(8)))  int      v8i;
typedef __attribute__((ext_vector_type(4)))  int      v4i;

union Frag16 { v16h h; unsigned u[8]; };

#define N_TOK 4096
#define C_DIM 512
#define HEADS 8
#define DH    64
#define LSTR  36   // LDS row stride in dwords for 64-half rows (32 data + 4 pad, 16B aligned)

// ---- CDNA5 wait helpers ---------------------------------------------------
__device__ __forceinline__ void wait_async0() {
#if __has_builtin(__builtin_amdgcn_s_wait_asynccnt)
    __builtin_amdgcn_s_wait_asynccnt(0);
#else
    asm volatile("s_wait_asynccnt 0x0" ::: "memory");
#endif
}
__device__ __forceinline__ void wait_tensor0() {
#if __has_builtin(__builtin_amdgcn_s_wait_tensorcnt)
    __builtin_amdgcn_s_wait_tensorcnt(0);
#else
    asm volatile("s_wait_tensorcnt 0x0" ::: "memory");
#endif
}
// per-lane async copy: 16 bytes global -> LDS (GV mode, ASYNCcnt-tracked)
__device__ __forceinline__ void async_b128(unsigned lds_off, const void* g) {
    asm volatile("global_load_async_to_lds_b128 %0, %1, off"
                 :: "v"(lds_off), "v"((unsigned long long)(size_t)g)
                 : "memory");
}

// ---- fragment loader: row-major 16-bit matrix in LDS ----------------------
// A-layout (16x32 f16): lane L holds row (L&15); dword d holds K-pair
//   kp = d + (d&4) + 4*(L>>4).  B-layout identical with "row" = column index,
// so D = X * Y^T loads rows of X (A frags) and rows of Y (B frags).
__device__ __forceinline__ v16h load_frag_lds(const unsigned* lds, int row_base,
                                              int stride_dw, int kdw_base, int lane) {
    Frag16 f;
    const int row = row_base + (lane & 15);
    const int hi4 = (lane >> 4) << 2;
    const unsigned* p = lds + row * stride_dw + kdw_base;
#pragma unroll
    for (int d = 0; d < 8; ++d) f.u[d] = p[d + (d & 4) + hi4];
    return f.h;
}

__device__ __forceinline__ float rowmax16(float v) {
#pragma unroll
    for (int m = 8; m >= 1; m >>= 1) v = fmaxf(v, __shfl_xor(v, m, 32));
    return v;
}
__device__ __forceinline__ float rowsum16(float v) {
#pragma unroll
    for (int m = 8; m >= 1; m >>= 1) v += __shfl_xor(v, m, 32);
    return v;
}

// ---- TDM load of a 64x64 f16 tile into LDS with 36-dword row stride -------
// One issue per block (wave 0).  D# per CDNA5 ISA ch.8: group0 {count, lds,
// gaddr, type=2}; group1 {data_size=2B, pad 4dw every 32dw, dims/strides}.
__device__ __forceinline__ void load_tile_64x64_f16(unsigned lds_base, const void* gsrc,
                                                    int tid, int wv) {
#if __has_builtin(__builtin_amdgcn_tensor_load_to_lds)
    if (wv == 0) {
        unsigned long long ga = (unsigned long long)(size_t)gsrc;
        v4u g0 = { 1u,                                   // count=1 (valid descriptor)
                   lds_base,                             // lds_addr (bytes)
                   (unsigned)ga,                         // global_addr[31:0]
                   (unsigned)((ga >> 32) & 0x1ffffffull) | 0x80000000u };  // [56:32] | type=2
        v8i g1;
        g1[0] = (int)((1u << 16)      // data_size = 2 bytes
                    | (1u << 20)      // pad_enable
                    | (4u << 22)      // pad_interval: 2^(4+1)=32 dwords
                    | (3u << 25));    // pad_amount: 3+1 = 4 dwords -> 36-dword rows
        g1[1] = (int)(64u << 16);     // tensor_dim0 = 64 (lo16)
        g1[2] = (int)(64u << 16);     // tensor_dim0 hi=0 | tensor_dim1 = 64 (lo16)
        g1[3] = (int)(64u << 16);     // tensor_dim1 hi=0 | tile_dim0 = 64
        g1[4] = (int)(64u);           // tile_dim1 = 64 | tile_dim2 = 0
        g1[5] = (int)(64u);           // tensor_dim0_stride = 64 (lo32)
        g1[6] = 0;                    // stride0 hi | tensor_dim1_stride lo = 0
        g1[7] = 0;
        v4i z4 = { 0, 0, 0, 0 };
        v8i z8 = { 0, 0, 0, 0, 0, 0, 0, 0 };
        __builtin_amdgcn_tensor_load_to_lds(g0, g1, z4, z4, z8, 0);
        wait_tensor0();
    }
    (void)tid;
#else
    // fallback: per-lane async B128 copies (64 rows x 128B, padded LDS rows)
#pragma unroll
    for (int i = 0; i < 2; ++i) {
        int lin = i * 256 + tid;
        int row = lin >> 3, seg = lin & 7;
        async_b128(lds_base + row * (LSTR * 4) + seg * 16,
                   (const char*)gsrc + row * (DH * 2) + seg * 16);
    }
    wait_async0();
    (void)wv;
#endif
}

// ===========================================================================
// Kernel 0: fp32 -> f16 conversion (x, w_qkv, w_out), vectorized.
// ===========================================================================
__global__ __launch_bounds__(256) void cvt_f16_kernel(const float* __restrict__ src,
                                                      _Float16* __restrict__ dst) {
    int i = blockIdx.x * 256 + threadIdx.x;
    float4 v = ((const float4*)src)[i];
    dst[i * 4 + 0] = (_Float16)v.x;
    dst[i * 4 + 1] = (_Float16)v.y;
    dst[i * 4 + 2] = (_Float16)v.z;
    dst[i * 4 + 3] = (_Float16)v.w;
}

// ===========================================================================
// Kernel 1: QKV projection.  qkv[t,o] = sum_c xh[c,t] * wqh[o,c]
// Tile: 128 tokens x 64 outputs, K-chunks of 64, 8 waves.
// A staged with in-LDS transpose (dword recombine); B staged via async B128.
// ===========================================================================
__global__ __launch_bounds__(256) void qkv_gemm_kernel(
    const _Float16* __restrict__ xh, const _Float16* __restrict__ wqh,
    _Float16* __restrict__ qb, _Float16* __restrict__ kb, _Float16* __restrict__ vb)
{
    __shared__ __align__(16) unsigned Ald[128 * LSTR];
    __shared__ __align__(16) unsigned Bld[64 * LSTR];

    const int tid  = threadIdx.x;
    const int lane = tid & 31;
    const int wv   = tid >> 5;
    const int t0   = (blockIdx.x % 32) * 128;
    const int o0   = (blockIdx.x / 32) * 64;           // 1536/64 = 24 tiles
    const unsigned bld_base = (unsigned)(size_t)Bld;
    const unsigned* Xd = (const unsigned*)xh;          // dword view: [C][N/2]

    v8f acc[4];
#pragma unroll
    for (int g = 0; g < 4; ++g) acc[g] = {};

    for (int c0 = 0; c0 < C_DIM; c0 += 64) {
        __syncthreads();
        // B tile: 64 rows x 64 ch = 128B/row, direct async copies
#pragma unroll
        for (int i = 0; i < 2; ++i) {
            int lin = i * 256 + tid;
            int row = lin >> 3, seg = lin & 7;
            async_b128(bld_base + row * (LSTR * 4) + seg * 16,
                       (const char*)(wqh + (size_t)(o0 + row) * C_DIM + c0) + seg * 16);
        }
        // A tile: transpose xh[c][t] -> Ald[t][c]; recombine dwords
#pragma unroll
        for (int i = 0; i < 8; ++i) {
            int lin = i * 256 + tid;
            int cp = lin >> 6;                  // channel pair 0..31
            int td = lin & 63;                  // token-dword 0..63 (2 tokens)
            int c  = c0 + 2 * cp;
            unsigned u0 = Xd[(size_t)c * (N_TOK / 2) + (t0 >> 1) + td];
            unsigned u1 = Xd[(size_t)(c + 1) * (N_TOK / 2) + (t0 >> 1) + td];
            int t = 2 * td;
            Ald[t * LSTR + cp]       = (u0 & 0xffffu) | (u1 << 16);
            Ald[(t + 1) * LSTR + cp] = (u0 >> 16) | (u1 & 0xffff0000u);
        }
        wait_async0();
        __syncthreads();

#pragma unroll
        for (int kc = 0; kc < 2; ++kc) {
            v16h a = load_frag_lds(Ald, wv * 16, LSTR, kc * 16, lane);
#pragma unroll
            for (int g = 0; g < 4; ++g) {
                v16h b = load_frag_lds(Bld, g * 16, LSTR, kc * 16, lane);
                acc[g] = __builtin_amdgcn_wmma_f32_16x16x32_f16(
                    false, a, false, b, (short)0, acc[g], false, false);
            }
        }
    }

    // epilogue: C-layout  row M = r + 8*(lane>>4), col = lane&15
    const int mhi = (lane >> 4) * 8;
    const int col = lane & 15;
#pragma unroll
    for (int g = 0; g < 4; ++g) {
#pragma unroll
        for (int r = 0; r < 8; ++r) {
            int t = t0 + wv * 16 + r + mhi;
            int o = o0 + g * 16 + col;
            float val = acc[g][r];
            int head = (o & 511) >> 6;
            int d    = o & 63;
            size_t idx = (size_t)head * N_TOK * DH + (size_t)t * DH + d;
            if (o < 512)        qb[idx] = (_Float16)(val * 8.0f);  // bake scale=sqrt(dh)
            else if (o < 1024)  kb[idx] = (_Float16)val;
            else                vb[idx] = (_Float16)val;
        }
    }
}

// ===========================================================================
// Kernel 2: flash attention.  Block = 1 head x 128 query rows (8 waves).
// K tile via TDM (or async fallback); V staged transposed; online softmax;
// P re-laid out through per-wave LDS for the P*V WMMAs.
// ===========================================================================
__global__ __launch_bounds__(256) void attn_kernel(
    const _Float16* __restrict__ Q, const _Float16* __restrict__ K,
    const _Float16* __restrict__ V, _Float16* __restrict__ O)
{
    __shared__ __align__(16) unsigned Kld[64 * LSTR];       // K tile [kv][dh]
    __shared__ __align__(16) unsigned Vld[64 * LSTR];       // V^T tile [dh][kv]
    __shared__ __align__(16) unsigned Pld[8 * 16 * LSTR];   // per-wave P [16 q][64 kv]

    const int tid  = threadIdx.x;
    const int lane = tid & 31;
    const int wv   = tid >> 5;
    const int head = blockIdx.x >> 5;
    const int q0   = (blockIdx.x & 31) * 128;

    const _Float16* Qh = Q + (size_t)head * N_TOK * DH;
    const _Float16* Kh = K + (size_t)head * N_TOK * DH;
    const unsigned short* Vh16 = (const unsigned short*)(V + (size_t)head * N_TOK * DH);
    const unsigned kld_base = (unsigned)(size_t)Kld;

    // Q fragments (2 K-chunks of 32 over dh=64): per-lane global loads
    Frag16 qa[2];
    {
        const int trow = q0 + wv * 16 + (lane & 15);
        const unsigned* qrow = (const unsigned*)(Qh + (size_t)trow * DH);
        const int hi4 = (lane >> 4) << 2;
#pragma unroll
        for (int c = 0; c < 2; ++c)
#pragma unroll
            for (int d = 0; d < 8; ++d)
                qa[c].u[d] = qrow[c * 16 + d + (d & 4) + hi4];
    }

    float m_run[8], l_run[8];
    v8f   oacc[4];
#pragma unroll
    for (int r = 0; r < 8; ++r) { m_run[r] = -3.0e38f; l_run[r] = 0.0f; }
#pragma unroll
    for (int g = 0; g < 4; ++g) oacc[g] = {};

    unsigned* Pw  = Pld + wv * 16 * LSTR;
    _Float16* Pwh = (_Float16*)Pw;
    const int mhi = (lane >> 4) * 8;
    const int col = lane & 15;

    for (int j0 = 0; j0 < N_TOK; j0 += 64) {
        __syncthreads();
        // K tile: TDM (wave 0) or async-copy fallback, padded to LSTR rows
        load_tile_64x64_f16(kld_base, Kh + (size_t)j0 * DH, tid, wv);
        // V^T tile: gather two kv rows per dword, store Vld[dd][kvpair]
#pragma unroll
        for (int i = 0; i < 8; ++i) {
            int lin = i * 256 + tid;
            int dd  = lin & 63;                 // dh index (coalesced across tid)
            int kvp = lin >> 6;                 // kv pair 0..31
            int kv  = j0 + 2 * kvp;
            unsigned a = Vh16[(size_t)kv * DH + dd];
            unsigned b = Vh16[(size_t)(kv + 1) * DH + dd];
            Vld[dd * LSTR + kvp] = a | (b << 16);
        }
        __syncthreads();

        // S = Q * K^T  (4 kv16 groups x 2 K-chunks)
        v8f s[4];
#pragma unroll
        for (int g = 0; g < 4; ++g) {
            v8f sc = {};
#pragma unroll
            for (int c = 0; c < 2; ++c) {
                v16h b = load_frag_lds(Kld, g * 16, LSTR, c * 16, lane);
                sc = __builtin_amdgcn_wmma_f32_16x16x32_f16(
                    false, qa[c].h, false, b, (short)0, sc, false, false);
            }
            s[g] = sc;
        }

        // online softmax
        float mnew[8], scl[8], lb[8];
#pragma unroll
        for (int r = 0; r < 8; ++r) {
            float mb = fmaxf(fmaxf(s[0][r], s[1][r]), fmaxf(s[2][r], s[3][r]));
            mb = rowmax16(mb);
            mnew[r] = fmaxf(m_run[r], mb);
            scl[r]  = __expf(m_run[r] - mnew[r]);
            m_run[r] = mnew[r];
            lb[r] = 0.0f;
        }
#pragma unroll
        for (int g = 0; g < 4; ++g)
#pragma unroll
            for (int r = 0; r < 8; ++r) {
                float p = __expf(s[g][r] - mnew[r]);
                lb[r] += p;
                Pwh[(r + mhi) * (LSTR * 2) + g * 16 + col] = (_Float16)p;
            }
#pragma unroll
        for (int r = 0; r < 8; ++r) l_run[r] = l_run[r] * scl[r] + rowsum16(lb[r]);
#pragma unroll
        for (int g = 0; g < 4; ++g)
#pragma unroll
            for (int r = 0; r < 8; ++r) oacc[g][r] *= scl[r];

        // O += P * V  (A = P from per-wave LDS, B = V^T rows)
#pragma unroll
        for (int g = 0; g < 4; ++g) {
#pragma unroll
            for (int c = 0; c < 2; ++c) {
                v16h pa  = load_frag_lds(Pw, 0, LSTR, c * 16, lane);
                v16h vbf = load_frag_lds(Vld, g * 16, LSTR, c * 16, lane);
                oacc[g] = __builtin_amdgcn_wmma_f32_16x16x32_f16(
                    false, pa, false, vbf, (short)0, oacc[g], false, false);
            }
        }
    }

    // finalize: O /= l, write token-major [t][C] f16
#pragma unroll
    for (int r = 0; r < 8; ++r) l_run[r] = 1.0f / l_run[r];
#pragma unroll
    for (int g = 0; g < 4; ++g)
#pragma unroll
        for (int r = 0; r < 8; ++r) {
            int t = q0 + wv * 16 + r + mhi;
            int c = head * DH + g * 16 + col;
            O[(size_t)t * C_DIM + c] = (_Float16)(oacc[g][r] * l_run[r]);
        }
}

// ===========================================================================
// Kernel 3: out projection.  out[o,t] = sum_c Ob[t,c] * woh[o,c], fp32 out
// in the reference's [C][N] layout.  Both tiles staged via async B128.
// ===========================================================================
__global__ __launch_bounds__(256) void out_gemm_kernel(
    const _Float16* __restrict__ Ob, const _Float16* __restrict__ woh,
    float* __restrict__ out)
{
    __shared__ __align__(16) unsigned Ald[128 * LSTR];
    __shared__ __align__(16) unsigned Bld[64 * LSTR];

    const int tid  = threadIdx.x;
    const int lane = tid & 31;
    const int wv   = tid >> 5;
    const int t0   = (blockIdx.x & 31) * 128;
    const int o0   = (blockIdx.x >> 5) * 64;           // 512/64 = 8 tiles
    const unsigned ald_base = (unsigned)(size_t)Ald;
    const unsigned bld_base = (unsigned)(size_t)Bld;

    v8f acc[4];
#pragma unroll
    for (int g = 0; g < 4; ++g) acc[g] = {};

    for (int c0 = 0; c0 < C_DIM; c0 += 64) {
        __syncthreads();
        // A tile: 128 rows x 128B, direct async copies
#pragma unroll
        for (int i = 0; i < 4; ++i) {
            int lin = i * 256 + tid;
            int row = lin >> 3, seg = lin & 7;
            async_b128(ald_base + row * (LSTR * 4) + seg * 16,
                       (const char*)(Ob + (size_t)(t0 + row) * C_DIM + c0) + seg * 16);
        }
        // B tile: 64 rows x 128B, direct async copies
#pragma unroll
        for (int i = 0; i < 2; ++i) {
            int lin = i * 256 + tid;
            int row = lin >> 3, seg = lin & 7;
            async_b128(bld_base + row * (LSTR * 4) + seg * 16,
                       (const char*)(woh + (size_t)(o0 + row) * C_DIM + c0) + seg * 16);
        }
        wait_async0();
        __syncthreads();

#pragma unroll
        for (int kc = 0; kc < 2; ++kc) {
            v16h a = load_frag_lds(Ald, wv * 16, LSTR, kc * 16, lane);
#pragma unroll
            for (int g = 0; g < 4; ++g) {
                v16h b = load_frag_lds(Bld, g * 16, LSTR, kc * 16, lane);
                acc[g] = __builtin_amdgcn_wmma_f32_16x16x32_f16(
                    false, a, false, b, (short)0, acc[g], false, false);
            }
        }
    }

    const int mhi = (lane >> 4) * 8;
    const int col = lane & 15;
#pragma unroll
    for (int g = 0; g < 4; ++g)
#pragma unroll
        for (int r = 0; r < 8; ++r) {
            int t = t0 + wv * 16 + r + mhi;
            int o = o0 + g * 16 + col;
            out[(size_t)o * N_TOK + t] = acc[g][r];
        }
}

// ===========================================================================
extern "C" void kernel_launch(void* const* d_in, const int* in_sizes, int n_in,
                              void* d_out, int out_size, void* d_ws, size_t ws_size,
                              hipStream_t stream) {
    (void)in_sizes; (void)n_in; (void)out_size; (void)ws_size;
    const float* x     = (const float*)d_in[0];
    const float* w_qkv = (const float*)d_in[1];
    const float* w_out = (const float*)d_in[2];
    float* out = (float*)d_out;

    // workspace layout (f16 elements)
    _Float16* xh  = (_Float16*)d_ws;                       // [512][4096]
    _Float16* wqh = xh  + (size_t)C_DIM * N_TOK;           // [1536][512]
    _Float16* woh = wqh + (size_t)3 * C_DIM * C_DIM;       // [512][512]
    _Float16* qb  = woh + (size_t)C_DIM * C_DIM;           // [8][4096][64]
    _Float16* kb  = qb  + (size_t)HEADS * N_TOK * DH;
    _Float16* vb  = kb  + (size_t)HEADS * N_TOK * DH;
    _Float16* ob  = vb  + (size_t)HEADS * N_TOK * DH;      // [4096][512]

    // 0) fp32 -> f16 pre-conversion
    cvt_f16_kernel<<<dim3((C_DIM * N_TOK) / 1024), dim3(256), 0, stream>>>(x, xh);
    cvt_f16_kernel<<<dim3((3 * C_DIM * C_DIM) / 1024), dim3(256), 0, stream>>>(w_qkv, wqh);
    cvt_f16_kernel<<<dim3((C_DIM * C_DIM) / 1024), dim3(256), 0, stream>>>(w_out, woh);
    // 1) QKV projection: 32 token tiles x 24 output tiles
    qkv_gemm_kernel<<<dim3(32 * 24), dim3(256), 0, stream>>>(xh, wqh, qb, kb, vb);
    // 2) fused flash attention: 8 heads x 32 query blocks
    attn_kernel<<<dim3(HEADS * 32), dim3(256), 0, stream>>>(qb, kb, vb, ob);
    // 3) output projection: 32 token tiles x 8 output tiles
    out_gemm_kernel<<<dim3(32 * 8), dim3(256), 0, stream>>>(ob, woh, out);
}
